// QLinearPacked_21139829031278
// MI455X (gfx1250) — matrix operand order repaired and checked by reading, not compiled, not run on hardware
//
#include <hip/hip_runtime.h>
#include <hip/hip_bf16.h>

#define IN_F   4096
#define OUT_F  14336
#define GROUP  16
#define KSLICES 4                     // K split across 4 waves, reduced via LDS
#define KSLICE_LEN (IN_F / KSLICES)   // 1024

typedef __attribute__((ext_vector_type(16))) __bf16 v16bf;
typedef __attribute__((ext_vector_type(8)))  float  v8f;

union BfTile {
    v16bf v;
    unsigned int d[8];
    uint4 q[2];
};

// e8m0 exponent combined with the fixed /64 mantissa scaling: 2^(e-127)/64 = 2^(e-133)
__device__ __forceinline__ float e8m0_div64(unsigned e) {
    return __uint_as_float(((e - 6u) & 0xFFu) << 23);
}

// 16 int8 mantissas -> 16 bf16 values scaled by sc.
// XOR-bias + cvt_f32_ubyteN + dual fma(u, sc, -128*sc); bf16 pack via v_perm_b32.
// Exact: int8 * 2^k has <=8 significand bits, so f32->bf16 truncation is lossless.
__device__ __forceinline__ v16bf dequant16(uint4 qw, float sc) {
    const float nb = -128.0f * sc;
    unsigned int dw[4] = {qw.x ^ 0x80808080u, qw.y ^ 0x80808080u,
                          qw.z ^ 0x80808080u, qw.w ^ 0x80808080u};
    BfTile b;
#pragma unroll
    for (int i = 0; i < 4; ++i) {
        unsigned int d = dw[i];
        float f0 = fmaf((float)(d & 0xFFu),         sc, nb);   // v_cvt_f32_ubyte0
        float f1 = fmaf((float)((d >> 8) & 0xFFu),  sc, nb);   // v_cvt_f32_ubyte1
        float f2 = fmaf((float)((d >> 16) & 0xFFu), sc, nb);   // v_cvt_f32_ubyte2
        float f3 = fmaf((float)(d >> 24),           sc, nb);   // v_cvt_f32_ubyte3
        // dst = { bf16(f_hi), bf16(f_lo) } : bytes [3:2] of each f32
        b.d[2 * i]     = __builtin_amdgcn_perm(__float_as_uint(f1), __float_as_uint(f0), 0x07060302u);
        b.d[2 * i + 1] = __builtin_amdgcn_perm(__float_as_uint(f3), __float_as_uint(f2), 0x07060302u);
    }
    return b.v;
}

__global__ __launch_bounds__(256) void qlinear_wmma_bf16(
    const __hip_bfloat16* __restrict__ input,      // [32, 4096]
    const unsigned char*  __restrict__ w_q,        // [14336*4096] int8 mantissa bits
    const unsigned char*  __restrict__ w_scales,   // [14336*4096/16] e8m0
    const unsigned char*  __restrict__ bias_q,     // [14336]
    const unsigned char*  __restrict__ bias_scales,// [14336/16]
    __hip_bfloat16*       __restrict__ out)        // [32, 14336]
{
    // 8 waves/block: 2 N-tiles x 4 K-slices. Partial sums reduced via LDS.
    __shared__ v8f red[8][2][32];   // 16 KB of the 320 KB WGP LDS

    const int lane   = threadIdx.x & 31;
    const int wave   = threadIdx.x >> 5;
    const int kslice = wave & 3;
    const int ntile  = blockIdx.x * 2 + (wave >> 2);
    const int n0     = ntile * 16;
    const int k0     = kslice * KSLICE_LEN;

    const int half = lane >> 4;   // K sub-block selector (CDNA5 wave32 striping)
    const int col  = lane & 15;   // N within tile (B/C/D), M within tile (A)

    const int out_row = n0 + col; // weight row == output column handled by this lane

    // B (weights): lane's 16 K-values = 16 consecutive bytes of one weight row
    // = exactly one MXINT8 group -> one b128 load per K-step.
    const unsigned char* wp = w_q + (size_t)out_row * IN_F + (size_t)half * GROUP + k0;
    // Scale base WITHOUT the half offset: 8B-aligned, one b64 covers 4 K-steps
    // for both halves (lane uses bytes half+{0,2} of each dword).
    const unsigned char* sp = w_scales + (size_t)out_row * (IN_F / GROUP) + (k0 / GROUP);

    // A (activations): 16x32 bf16 tile; lane holds row `col`, K-chunks
    // {16*half..+7} and {16*half+8..+7} -> two 16B loads per tile.
    const __hip_bfloat16* a0p = input + (size_t)col        * IN_F + half * 16 + k0;  // rows 0-15
    const __hip_bfloat16* a1p = input + (size_t)(col + 16) * IN_F + half * 16 + k0;  // rows 16-31

    v8f acc0 = {};  // output rows 0..15 (partial over this K-slice)
    v8f acc1 = {};  // output rows 16..31

    const int hsh = 8 * half;   // byte shift selecting this half's scale

    for (int k = 0; k < KSLICE_LEN; k += 128) {
        // 8 e8m0 scale bytes = 8 quant groups = 4 K-steps for both halves
        uint2 se = *(const uint2*)(sp + (k >> 4));

#pragma unroll
        for (int j = 0; j < 4; ++j) {
            const int kk = k + 32 * j;

            __builtin_prefetch(wp + kk + 256, 0, 1);   // stream weights ahead

            uint4 qw = *(const uint4*)(wp + kk);

            unsigned sdw = (j < 2) ? se.x : se.y;
            unsigned e   = (sdw >> (hsh + 16 * (j & 1))) & 0xFFu;

            BfTile a0, a1;
            a0.q[0] = *(const uint4*)(a0p + kk);
            a0.q[1] = *(const uint4*)(a0p + kk + 8);
            a1.q[0] = *(const uint4*)(a1p + kk);
            a1.q[1] = *(const uint4*)(a1p + kk + 8);

            v16bf b = dequant16(qw, e8m0_div64(e));

            acc0 = __builtin_amdgcn_wmma_f32_16x16x32_bf16(
                false, a0.v, false, b, (short)0, acc0, false, false);
            acc1 = __builtin_amdgcn_wmma_f32_16x16x32_bf16(
                false, a1.v, false, b, (short)0, acc1, false, false);
        }
    }

    // ---- cross-wave K reduction through LDS ----
    if (kslice != 0) {
        red[wave][0][lane] = acc0;
        red[wave][1][lane] = acc1;
    }
    __syncthreads();

    if (kslice == 0) {
#pragma unroll
        for (int s = 1; s < KSLICES; ++s) {
            acc0 += red[wave + s][0][lane];
            acc1 += red[wave + s][1][lane];
        }

        // Bias: column n = out_row; its group index is exactly `ntile`.
        float bsc = e8m0_div64((unsigned)bias_scales[out_row >> 4]);
        float bv  = (float)(signed char)bias_q[out_row] * bsc;

        // C/D layout: VGPR r of lane (half,col) holds row M = r + 8*half, col N = col.
#pragma unroll
        for (int r = 0; r < 8; ++r) {
            int m = r + 8 * half;
            out[(size_t)m        * OUT_F + out_row] = __float2bfloat16(acc0[r] + bv);
            out[(size_t)(m + 16) * OUT_F + out_row] = __float2bfloat16(acc1[r] + bv);
        }
    }
}

extern "C" void kernel_launch(void* const* d_in, const int* in_sizes, int n_in,
                              void* d_out, int out_size, void* d_ws, size_t ws_size,
                              hipStream_t stream) {
    const __hip_bfloat16* input       = (const __hip_bfloat16*)d_in[0];
    const unsigned char*  w_q         = (const unsigned char*)d_in[1];
    const unsigned char*  w_scales    = (const unsigned char*)d_in[2];
    const unsigned char*  bias_q      = (const unsigned char*)d_in[3];
    const unsigned char*  bias_scales = (const unsigned char*)d_in[4];
    __hip_bfloat16*       out         = (__hip_bfloat16*)d_out;

    dim3 grid(OUT_F / 16 / 2);   // 896 N-tiles, 2 per block -> 448 blocks, 8 waves each
    dim3 block(256);
    hipLaunchKernelGGL(qlinear_wmma_bf16, grid, block, 0, stream,
                       input, w_q, w_scales, bias_q, bias_scales, out);
}
